// ScaledDotProductAttention_13529146983095
// MI455X (gfx1250) — matrix operand order repaired and checked by reading, not compiled, not run on hardware
//
#include <hip/hip_runtime.h>
#include <math.h>

// ---- CDNA5 (gfx1250) wave32 WMMA fused attention ----
typedef __attribute__((ext_vector_type(16))) _Float16 v16h;
typedef __attribute__((ext_vector_type(8)))  _Float16 v8h;
typedef __attribute__((ext_vector_type(4)))  _Float16 v4h;
typedef __attribute__((ext_vector_type(8)))  float    v8f;

#define HEAD_DIM 64
#define LQ_TILE  64      // queries per workgroup (4 waves x 16 rows)
#define KCHUNK   64      // keys per LDS chunk
#define PAD      8
#define QS (HEAD_DIM + PAD)   // 72 halves = 144B row stride (16B multiple)
#define KS (HEAD_DIM + PAD)
#define VS (KCHUNK  + PAD)
#define PS (KCHUNK  + PAD)

#define B_DIM 16
#define LQ    2048
#define LK    2048
#define SCALE 0.125f     // 1/sqrt(64)

__device__ __forceinline__ v16h ld_frag(const _Float16* p0, const _Float16* p1) {
  union { v16h v; v8h h[2]; } u;
  u.h[0] = *(const v8h*)p0;   // ds_load_b128
  u.h[1] = *(const v8h*)p1;   // ds_load_b128
  return u.v;
}

__device__ __forceinline__ v8f wmma_ff16(v16h a, v16h b, v8f c) {
  return __builtin_amdgcn_wmma_f32_16x16x32_f16(false, a, false, b, (short)0, c,
                                                false, false);
}

// reductions across the 16-lane halves (matches C/D layout: lane = column)
__device__ __forceinline__ float rmax16(float x) {
  x = fmaxf(x, __shfl_xor(x, 1, 32));
  x = fmaxf(x, __shfl_xor(x, 2, 32));
  x = fmaxf(x, __shfl_xor(x, 4, 32));
  x = fmaxf(x, __shfl_xor(x, 8, 32));
  return x;
}
__device__ __forceinline__ float rsum16(float x) {
  x += __shfl_xor(x, 1, 32);
  x += __shfl_xor(x, 2, 32);
  x += __shfl_xor(x, 4, 32);
  x += __shfl_xor(x, 8, 32);
  return x;
}

// convert 8 consecutive fp32 (two float4) to one b128 LDS store
__device__ __forceinline__ void cvt_store8(_Float16* dst, float4 a, float4 b,
                                           float scale) {
  v8h h;
  h[0] = (_Float16)(a.x * scale); h[1] = (_Float16)(a.y * scale);
  h[2] = (_Float16)(a.z * scale); h[3] = (_Float16)(a.w * scale);
  h[4] = (_Float16)(b.x * scale); h[5] = (_Float16)(b.y * scale);
  h[6] = (_Float16)(b.z * scale); h[7] = (_Float16)(b.w * scale);
  *(v8h*)dst = h;                 // ds_store_b128
}

__global__ __launch_bounds__(128)
void fa_fused_kernel(const float* __restrict__ Q, const float* __restrict__ K,
                     const float* __restrict__ V, float* __restrict__ Ctx,
                     float* __restrict__ Attn)
{
  __shared__ alignas(16) _Float16 sQ[LQ_TILE][QS];
  __shared__ alignas(16) _Float16 sK[KCHUNK][KS];
  __shared__ alignas(16) _Float16 sVt[HEAD_DIM][VS];   // V transposed: [d][key]
  __shared__ alignas(16) _Float16 sP[4][16][PS];       // per-wave P scratch

  const int b     = blockIdx.y;
  const int qtile = blockIdx.x;
  const int tid   = threadIdx.x;
  const int lane  = tid & 31;
  const int wave  = tid >> 5;
  const int lo    = lane & 15;
  const int grp   = lane >> 4;

  const float* Qb = Q + ((size_t)b * LQ + (size_t)qtile * LQ_TILE) * HEAD_DIM;
  const float* Kb = K + (size_t)b * LK * HEAD_DIM;
  const float* Vb = V + (size_t)b * LK * HEAD_DIM;

  const int lr  = tid >> 1;          // loader row 0..63
  const int ld0 = (tid & 1) * 32;    // loader col base 0/32

  // ---- stage Q tile (pre-scaled by 1/sqrt(d)) into LDS as f16 ----
  {
    const float4* src = (const float4*)(Qb + lr * HEAD_DIM + ld0);
#pragma unroll
    for (int j = 0; j < 4; ++j)
      cvt_store8(&sQ[lr][ld0 + j * 8], src[2 * j], src[2 * j + 1], SCALE);
  }
  __syncthreads();

  // ---- per-wave loop-invariant A fragments of Q (16 rows x 64 d) ----
  const int qr0 = wave * 16;
  v16h aQ[2];
#pragma unroll
  for (int ks = 0; ks < 2; ++ks)
    aQ[ks] = ld_frag(&sQ[qr0 + lo][ks * 32 + grp * 8],
                     &sQ[qr0 + lo][ks * 32 + 16 + grp * 8]);

  float m_run[8], l_run[8];
  v8f o[4];
#pragma unroll
  for (int v = 0; v < 8; ++v) { m_run[v] = -INFINITY; l_run[v] = 0.f; }
#pragma unroll
  for (int t = 0; t < 4; ++t) { v8f z = {}; o[t] = z; }

  // V-transpose staging geometry: 4x4 register blocks, two per thread
  const int bl0 = tid;               // block ids tid and tid+128 of 256
  // =================== Pass A: online softmax + P@V ===================
  for (int kc = 0; kc < LK; kc += KCHUNK) {
    __syncthreads();
    {
      // K chunk -> sK (row-major f16), b128 stores
      const float4* ksrc = (const float4*)(Kb + (size_t)(kc + lr) * HEAD_DIM + ld0);
#pragma unroll
      for (int j = 0; j < 4; ++j)
        cvt_store8(&sK[lr][ld0 + j * 8], ksrc[2 * j], ksrc[2 * j + 1], 1.0f);

      // V chunk -> sVt (transposed f16) via 4x4 register blocks, b64 stores
#pragma unroll
      for (int half = 0; half < 2; ++half) {
        const int bl = bl0 + half * 128;
        const int br = (bl >> 4) << 2;         // key-row base 0..60
        const int bc = (bl & 15) << 2;         // d base      0..60
        float rr[4][4];
#pragma unroll
        for (int i = 0; i < 4; ++i) {
          float4 f = *(const float4*)(Vb + (size_t)(kc + br + i) * HEAD_DIM + bc);
          rr[i][0] = f.x; rr[i][1] = f.y; rr[i][2] = f.z; rr[i][3] = f.w;
        }
#pragma unroll
        for (int j = 0; j < 4; ++j) {
          v4h h; h.x = (_Float16)rr[0][j]; h.y = (_Float16)rr[1][j];
                 h.z = (_Float16)rr[2][j]; h.w = (_Float16)rr[3][j];
          *(v4h*)(&sVt[bc + j][br]) = h;       // ds_store_b64
        }
      }
    }
    __syncthreads();

    // prefetch next chunk (global_prefetch_b8): 128 threads x 128B lines
    if (kc + KCHUNK < LK) {
      __builtin_prefetch(Kb + (size_t)(kc + KCHUNK + lr) * HEAD_DIM + ld0, 0, 3);
      __builtin_prefetch(Vb + (size_t)(kc + KCHUNK + lr) * HEAD_DIM + ld0, 0, 3);
    }

    // S = Q*K^T  (this wave: 16 rows x 64 keys = 4 tiles, K-dim 64 = 2 steps)
    v8f acc[4];
#pragma unroll
    for (int t = 0; t < 4; ++t) {
      v8f c = {};
#pragma unroll
      for (int ks = 0; ks < 2; ++ks) {
        v16h bK = ld_frag(&sK[t * 16 + lo][ks * 32 + grp * 8],
                          &sK[t * 16 + lo][ks * 32 + 16 + grp * 8]);
        c = wmma_ff16(aQ[ks], bK, c);
      }
      acc[t] = c;
    }

    // online softmax update (row v+8*grp per VGPR index v)
#pragma unroll
    for (int v = 0; v < 8; ++v) {
      float mx = fmaxf(fmaxf(acc[0][v], acc[1][v]), fmaxf(acc[2][v], acc[3][v]));
      mx = rmax16(mx);
      float nm = fmaxf(m_run[v], mx);
      float alpha = __expf(m_run[v] - nm);
      l_run[v] *= alpha;
#pragma unroll
      for (int t = 0; t < 4; ++t) o[t][v] *= alpha;
      m_run[v] = nm;
    }
#pragma unroll
    for (int v = 0; v < 8; ++v) {
      float s = 0.f;
#pragma unroll
      for (int t = 0; t < 4; ++t) {
        float p = __expf(acc[t][v] - m_run[v]);
        sP[wave][v + 8 * grp][t * 16 + lo] = (_Float16)p;   // accum->A relayout
        s += p;
      }
      l_run[v] += rsum16(s);
    }

    // O += P @ V  (contraction over 64 keys = 2 steps, 4 d-tiles)
#pragma unroll
    for (int ks = 0; ks < 2; ++ks) {
      v16h aP = ld_frag(&sP[wave][lo][ks * 32 + grp * 8],
                        &sP[wave][lo][ks * 32 + 16 + grp * 8]);
#pragma unroll
      for (int t = 0; t < 4; ++t) {
        v16h bV = ld_frag(&sVt[t * 16 + lo][ks * 32 + grp * 8],
                          &sVt[t * 16 + lo][ks * 32 + 16 + grp * 8]);
        o[t] = wmma_ff16(aP, bV, o[t]);
      }
    }
  }

  // ---- write context = O / l (streamed, written once) ----
  float inv_l[8];
#pragma unroll
  for (int v = 0; v < 8; ++v) inv_l[v] = 1.0f / l_run[v];
  {
    const size_t ob = ((size_t)b * LQ + (size_t)qtile * LQ_TILE + qr0) * HEAD_DIM;
#pragma unroll
    for (int v = 0; v < 8; ++v) {
      const int row = v + 8 * grp;
#pragma unroll
      for (int t = 0; t < 4; ++t)
        __builtin_nontemporal_store(o[t][v] * inv_l[v],
            &Ctx[ob + (size_t)row * HEAD_DIM + t * 16 + lo]);
    }
  }

  // ============ Pass B: recompute S, emit normalized attention ============
  // K is L2-resident (25 MB total QKV vs 192 MB L2) so the re-read is cheap;
  // attention (268 MB) is written exactly once, non-temporal.
  for (int kc = 0; kc < LK; kc += KCHUNK) {
    __syncthreads();
    {
      const float4* ksrc = (const float4*)(Kb + (size_t)(kc + lr) * HEAD_DIM + ld0);
#pragma unroll
      for (int j = 0; j < 4; ++j)
        cvt_store8(&sK[lr][ld0 + j * 8], ksrc[2 * j], ksrc[2 * j + 1], 1.0f);
    }
    __syncthreads();

    if (kc + KCHUNK < LK)
      __builtin_prefetch(Kb + (size_t)(kc + KCHUNK + lr) * HEAD_DIM + ld0, 0, 3);

#pragma unroll
    for (int t = 0; t < 4; ++t) {
      v8f c = {};
#pragma unroll
      for (int ks = 0; ks < 2; ++ks) {
        v16h bK = ld_frag(&sK[t * 16 + lo][ks * 32 + grp * 8],
                          &sK[t * 16 + lo][ks * 32 + 16 + grp * 8]);
        c = wmma_ff16(aQ[ks], bK, c);
      }
#pragma unroll
      for (int v = 0; v < 8; ++v) {
        float p = __expf(c[v] - m_run[v]) * inv_l[v];
        size_t idx = ((size_t)b * LQ + (size_t)qtile * LQ_TILE + qr0 + v + 8 * grp)
                         * (size_t)LK + kc + t * 16 + lo;
        __builtin_nontemporal_store(p, &Attn[idx]);   // lanes 0-15: 64B contiguous
      }
    }
  }
}

extern "C" void kernel_launch(void* const* d_in, const int* in_sizes, int n_in,
                              void* d_out, int out_size, void* d_ws, size_t ws_size,
                              hipStream_t stream) {
  (void)in_sizes; (void)n_in; (void)out_size; (void)d_ws; (void)ws_size;
  const float* q = (const float*)d_in[0];
  const float* k = (const float*)d_in[1];
  const float* v = (const float*)d_in[2];
  float* ctx  = (float*)d_out;                                   // [B, Lq, D]
  float* attn = (float*)d_out + (size_t)B_DIM * LQ * HEAD_DIM;   // [B, Lq, Lk]
  dim3 grid(LQ / LQ_TILE, B_DIM);   // (32, 16)
  dim3 block(128);                  // 4 wave32 waves
  fa_fused_kernel<<<grid, block, 0, stream>>>(q, k, v, ctx, attn);
}
